// PointTransformerFeatureExtractor_29970281791914
// MI455X (gfx1250) — matrix operand order repaired and checked by reading, not compiled, not run on hardware
//
#include <hip/hip_runtime.h>
#include <hip/hip_bf16.h>

// Point Transformer feature extractor for MI455X (gfx1250, wave32, WMMA).
// Heavy per-(point,neighbor) GEMMs run on v_wmma_f32_16x16x32_bf16 with
// LDS pre-swizzled into per-lane fragment layout (contiguous 32B loads,
// vector D-tile stores). Everything else stays in f32 VALU.

#define BATCH 8
#define NPTS  4096
#define KNN   16
#define NHEAD 4

typedef __bf16 bf16_t;
typedef __attribute__((ext_vector_type(16))) __bf16 v16bf;
typedef __attribute__((ext_vector_type(8)))  __bf16 v8bf;
typedef __attribute__((ext_vector_type(8)))  float  v8f;

static __device__ __forceinline__ bf16_t to_bf(float f) { return (bf16_t)f; }

// ---------------------------------------------------------------------------
// Kernel 1: center + unit-sphere normalize each cloud.  x: [B,3,N] -> pos [B,N,3]
// ---------------------------------------------------------------------------
__global__ __launch_bounds__(256) void k_normalize(const float* __restrict__ x,
                                                   float* __restrict__ pos) {
  const int b = blockIdx.x;
  const float* xb = x + (size_t)b * 3 * NPTS;
  __shared__ float red[256];
  __shared__ float cmean[3];

  float comp[3] = {0.f, 0.f, 0.f};
  for (int n = threadIdx.x; n < NPTS; n += 256) {
    comp[0] += xb[n];
    comp[1] += xb[NPTS + n];
    comp[2] += xb[2 * NPTS + n];
  }
  for (int c = 0; c < 3; ++c) {
    red[threadIdx.x] = comp[c];
    __syncthreads();
    for (int s = 128; s > 0; s >>= 1) {
      if (threadIdx.x < s) red[threadIdx.x] += red[threadIdx.x + s];
      __syncthreads();
    }
    if (threadIdx.x == 0) cmean[c] = red[0] * (1.0f / NPTS);
    __syncthreads();
  }
  const float cx = cmean[0], cy = cmean[1], cz = cmean[2];

  float mx = 0.f;
  for (int n = threadIdx.x; n < NPTS; n += 256) {
    float dx = xb[n] - cx, dy = xb[NPTS + n] - cy, dz = xb[2 * NPTS + n] - cz;
    mx = fmaxf(mx, sqrtf(dx * dx + dy * dy + dz * dz));
  }
  red[threadIdx.x] = mx;
  __syncthreads();
  for (int s = 128; s > 0; s >>= 1) {
    if (threadIdx.x < s) red[threadIdx.x] = fmaxf(red[threadIdx.x], red[threadIdx.x + s]);
    __syncthreads();
  }
  const float inv = 1.0f / (red[0] + 1e-8f);
  for (int n = threadIdx.x; n < NPTS; n += 256) {
    float* p = pos + ((size_t)b * NPTS + n) * 3;
    p[0] = (xb[n] - cx) * inv;
    p[1] = (xb[NPTS + n] - cy) * inv;
    p[2] = (xb[2 * NPTS + n] - cz) * inv;
  }
}

// ---------------------------------------------------------------------------
// Kernel 2: brute-force KNN (top-16 by -||pi-pj||^2) + pdiff precompute.
// ---------------------------------------------------------------------------
#define KCHUNK 512
__global__ __launch_bounds__(256) void k_knn(const float* __restrict__ pos,
                                             int* __restrict__ idx,
                                             float* __restrict__ pdiff) {
  const int b = blockIdx.y;
  const int n = blockIdx.x * 256 + threadIdx.x;
  const float* pb = pos + (size_t)b * NPTS * 3;
  const float px = pb[n * 3 + 0], py = pb[n * 3 + 1], pz = pb[n * 3 + 2];

  float bestd[KNN];
  int   besti[KNN];
#pragma unroll
  for (int i = 0; i < KNN; ++i) { bestd[i] = -1e30f; besti[i] = 0; }

  __shared__ float sp[KCHUNK * 3];
  for (int base = 0; base < NPTS; base += KCHUNK) {
    __syncthreads();
    for (int t = threadIdx.x; t < KCHUNK * 3; t += 256) sp[t] = pb[base * 3 + t];
    __syncthreads();
    for (int j = 0; j < KCHUNK; ++j) {
      float dx = px - sp[j * 3 + 0];
      float dy = py - sp[j * 3 + 1];
      float dz = pz - sp[j * 3 + 2];
      float v = -(dx * dx + dy * dy + dz * dz);
      if (v > bestd[KNN - 1]) {                 // strict '>' => lowest index wins ties
        int p = KNN - 1;
        while (p > 0 && bestd[p - 1] < v) {
          bestd[p] = bestd[p - 1];
          besti[p] = besti[p - 1];
          --p;
        }
        bestd[p] = v;
        besti[p] = base + j;
      }
    }
  }
  for (int k = 0; k < KNN; ++k) {
    const int j = besti[k];
    idx[((size_t)b * NPTS + n) * KNN + k] = j;
    float* pd = pdiff + (((size_t)b * NPTS + n) * KNN + k) * 3;
    pd[0] = px - pb[j * 3 + 0];
    pd[1] = py - pb[j * 3 + 1];
    pd[2] = pz - pb[j * 3 + 2];
  }
}

// ---------------------------------------------------------------------------
// Kernel 3: input embedding f = relu(pos @ in_W + in_b), 3->16.
// ---------------------------------------------------------------------------
__global__ __launch_bounds__(256) void k_embed(const float* __restrict__ pos,
                                               const float* __restrict__ W,
                                               const float* __restrict__ bvec,
                                               float* __restrict__ f) {
  const int gid = blockIdx.x * 256 + threadIdx.x;
  if (gid >= BATCH * NPTS) return;
  const float* p = pos + (size_t)gid * 3;
  const float x0 = p[0], x1 = p[1], x2 = p[2];
#pragma unroll
  for (int c = 0; c < 16; ++c) {
    float v = bvec[c] + x0 * W[c] + x1 * W[16 + c] + x2 * W[32 + c];
    f[(size_t)gid * 16 + c] = fmaxf(v, 0.f);
  }
}

// ---------------------------------------------------------------------------
// Fragment-layout helpers (wave32 WMMA 16x16x32, per CDNA5 ISA tables).
//  A: lane<16 holds M=lane, K in {0-7,16-23}; lane>=16 holds M=lane-16,
//     K in {8-15,24-31}. Element e of the v16bf: k = ((e>>3)<<4)+(half<<3)+(e&7).
//  B: column = lane&15; same k order as A: element e: k = ((e>>3)<<4)+(half<<3)+(e&7).
//  D: VGPR r <-> M = r + half*8; column = lane&15.
// We store sA/sW already swizzled so each lane's fragment is 16 contiguous
// bf16 (32 bytes -> 2x ds_load_b128).  Inverse swizzle (k -> half,e):
//     half = (k>>3)&1 ;  e = (k&7) + ((k>>4)<<3)
// ---------------------------------------------------------------------------

// ---------------------------------------------------------------------------
// Kernel 4: fused Point Transformer layer.
// Workgroup = 256 threads (8 wave32) handles 16 query points = 256 (n,k) rows.
// ---------------------------------------------------------------------------
__global__ __launch_bounds__(256, 1) void k_layer(
    const float* __restrict__ xin, int xin_stride, int cin,
    float* __restrict__ xout, int cout, int has_sc,
    const int* __restrict__ idx, const float* __restrict__ pdiff,
    const float* __restrict__ Wq,  const float* __restrict__ bq,
    const float* __restrict__ Wk,  const float* __restrict__ bk,
    const float* __restrict__ Wv,  const float* __restrict__ bv,
    const float* __restrict__ Wp1, const float* __restrict__ bp1,
    const float* __restrict__ Wp2, const float* __restrict__ bp2,
    const float* __restrict__ Wfc, const float* __restrict__ bfc,
    const float* __restrict__ Wsc, const float* __restrict__ bsc) {
  const int tid  = threadIdx.x;
  const int lane = tid & 31;
  const int wave = tid >> 5;
  const int half = lane >> 4;
  const int mrow = lane & 15;
  const int b  = blockIdx.y;
  const int n0 = blockIdx.x * 16;          // first point of this tile
  const int d  = cout / NHEAD;
  const float dinv = 1.0f / (float)d;
  const int ntz = (cout == 64) ? 2 : ((cout == 32) ? 1 : 0);
  const int ntn = 1 << ntz;                // N-tiles (cout/16)
  const int ntiles = 16 * ntn;             // 16 M-tiles x ntn N-tiles
  const int kchunks = (cout + 31) >> 5;    // K chunks of 32 for the pe GEMM

  // ~150 KB LDS (fits CU-mode 160 KB; spec allows 320 KB/WGP)
  __shared__ alignas(32) bf16_t sA[2 * 16 * 32 * 16]; // A fragments: [kc][mt][lane][e]
  __shared__ alignas(32) bf16_t sW[2 * 4 * 2 * 16 * 16]; // B frags: [kc][nt][half][col][e]
  __shared__ alignas(32) float  sVPE[64 * 256];  // column-major [col][row]: v, then v+pe
  __shared__ alignas(32) bf16_t sKPE[64 * 256];  // column-major [col][row]: -kk, then pe-kk
  __shared__ float  sQ[16 * 64];      // q, later reused as agg
  __shared__ float  sXin[16 * 32];    // input features of the 16 tile points
  __shared__ float  sPd[256 * 3];     // pdiff rows
  __shared__ float  sS[256 * 4];      // attention scores / weights
  __shared__ float  sMax[16 * 4];
  __shared__ float  sSum[16 * 4];
  __shared__ float  sBias[64];

  __builtin_prefetch(Wp2, 0, 1);      // global_prefetch_b8: warm caches for later pass
  __builtin_prefetch(Wfc, 0, 1);

  auto load_a = [&](int kc, int mt) -> v16bf {
    return *(const v16bf*)(sA + (((kc * 16 + mt) * 32 + lane) << 4));
  };
  auto load_b = [&](int kc, int nt) -> v16bf {
    return *(const v16bf*)(sW + (((((kc << ntz) + nt) * 2 + half) * 16 + mrow) << 4));
  };

  // ---- P0: stage inputs, gather neighbors (swizzled), stage Wk (swizzled) ----
  for (int e = tid; e < 16 * 32; e += 256) {
    int p = e >> 5, c = e & 31;
    sXin[e] = (c < cin) ? xin[((size_t)b * NPTS + n0 + p) * xin_stride + c] : 0.f;
  }
  {
    const size_t pdbase = ((size_t)b * NPTS + n0) * KNN * 3;
    for (int e = tid; e < 256 * 3; e += 256) sPd[e] = pdiff[pdbase + e];
  }
  for (int e = tid; e < 256 * 32; e += 256) {
    int row = e >> 5, j = e & 31;
    int p = row >> 4, kk2 = row & 15;
    int src = idx[((size_t)b * NPTS + n0 + p) * KNN + kk2];
    float v = (j < cin) ? xin[((size_t)b * NPTS + src) * xin_stride + j] : 0.f;
    int hh = (j >> 3) & 1;
    int ee = (j & 7) + ((j >> 4) << 3);
    sA[(((row >> 4) * 32 + ((hh << 4) | (row & 15))) << 4) + ee] = to_bf(v);
  }
  for (int e = tid; e < ntn * 512; e += 256) {   // dest-driven: e == b_idx(0,nt,hh,col,ee)
    int ee = e & 15, col = (e >> 4) & 15, hh = (e >> 8) & 1, nt = e >> 9;
    int k = ((ee >> 3) << 4) + hh * 8 + (ee & 7), c = nt * 16 + col;
    sW[e] = to_bf((k < cin) ? Wk[k * cout + c] : 0.f);
  }
  if (tid < 64) sBias[tid] = (tid < cout) ? bk[tid] : 0.f;
  __syncthreads();

  // ---- P1: q = x@Wq + bq (per-point, scalar) ----
  for (int e = tid; e < 16 * cout; e += 256) {
    int p = e / cout, c = e % cout;
    float acc = bq[c];
    for (int j2 = 0; j2 < cin; ++j2) acc += sXin[p * 32 + j2] * Wq[j2 * cout + c];
    sQ[p * 64 + c] = acc;
  }

  // ---- P1b: kk = xn@Wk + bk  -> sKPE = -(kk) (WMMA, K=32 padded) ----
  for (int tI = wave; tI < ntiles; tI += 8) {
    const int mt = tI & 15, nt = tI >> 4;
    const int ncol = nt * 16 + mrow;
    v16bf a = load_a(0, mt);
    v16bf bb = load_b(0, nt);
    v8f acc = {};
    acc = __builtin_amdgcn_wmma_f32_16x16x32_bf16(false, a, false, bb,
                                                  (short)0, acc, false, false);
    v8bf outp;
#pragma unroll
    for (int r = 0; r < 8; ++r) outp[r] = to_bf(-(acc[r] + sBias[ncol]));
    *(v8bf*)(sKPE + ncol * 256 + mt * 16 + half * 8) = outp;
  }
  __syncthreads();

  // ---- P2: stage Wv (swizzled) ----
  for (int e = tid; e < ntn * 512; e += 256) {
    int ee = e & 15, col = (e >> 4) & 15, hh = (e >> 8) & 1, nt = e >> 9;
    int k = ((ee >> 3) << 4) + hh * 8 + (ee & 7), c = nt * 16 + col;
    sW[e] = to_bf((k < cin) ? Wv[k * cout + c] : 0.f);
  }
  if (tid < 64) sBias[tid] = (tid < cout) ? bv[tid] : 0.f;
  __syncthreads();

  // ---- P3: v = xn@Wv + bv -> sVPE (WMMA) ----
  for (int tI = wave; tI < ntiles; tI += 8) {
    const int mt = tI & 15, nt = tI >> 4;
    const int ncol = nt * 16 + mrow;
    v16bf a = load_a(0, mt);
    v16bf bb = load_b(0, nt);
    v8f acc = {};
    acc = __builtin_amdgcn_wmma_f32_16x16x32_bf16(false, a, false, bb,
                                                  (short)0, acc, false, false);
    v8f vv;
#pragma unroll
    for (int r = 0; r < 8; ++r) vv[r] = acc[r] + sBias[ncol];
    *(v8f*)(sVPE + ncol * 256 + mt * 16 + half * 8) = vv;
  }
  __syncthreads();

  // ---- P4: t = relu(pdiff@Wp1 + bp1) into sA (swizzled); stage Wp2 (swizzled) ----
  for (int e = tid; e < 256 * kchunks * 32; e += 256) {
    const int jbits = (kchunks == 2) ? 6 : 5;
    int row = e >> jbits, j = e & ((kchunks << 5) - 1);
    float v = 0.f;
    if (j < cout) {
      v = bp1[j] + sPd[row * 3 + 0] * Wp1[j] + sPd[row * 3 + 1] * Wp1[cout + j] +
          sPd[row * 3 + 2] * Wp1[2 * cout + j];
      v = fmaxf(v, 0.f);
    }
    int kc = j >> 5, jl = j & 31;
    int hh = (jl >> 3) & 1;
    int ee = (jl & 7) + ((jl >> 4) << 3);
    sA[(((kc * 16 + (row >> 4)) * 32 + ((hh << 4) | (row & 15))) << 4) + ee] = to_bf(v);
  }
  for (int e = tid; e < kchunks * ntn * 512; e += 256) {
    int ee = e & 15, col = (e >> 4) & 15, hh = (e >> 8) & 1, q = e >> 9;
    int nt = q & (ntn - 1), kc = q >> ntz;
    int k = kc * 32 + ((ee >> 3) << 4) + hh * 8 + (ee & 7), c = nt * 16 + col;
    sW[e] = to_bf((k < cout) ? Wp2[k * cout + c] : 0.f);
  }
  if (tid < 64) sBias[tid] = (tid < cout) ? bp2[tid] : 0.f;
  __syncthreads();

  // ---- P5: pe = t@Wp2 + bp2; accumulate into pe-kk and v+pe (WMMA, K chunks) ----
  for (int tI = wave; tI < ntiles; tI += 8) {
    const int mt = tI & 15, nt = tI >> 4;
    const int ncol = nt * 16 + mrow;
    v8f acc = {};
    for (int kc = 0; kc < kchunks; ++kc) {
      v16bf a = load_a(kc, mt);
      v16bf bb = load_b(kc, nt);
      acc = __builtin_amdgcn_wmma_f32_16x16x32_bf16(false, a, false, bb,
                                                    (short)0, acc, false, false);
    }
    const int base = ncol * 256 + mt * 16 + half * 8;
    v8f vv = *(v8f*)(sVPE + base);
    v8bf kp = *(v8bf*)(sKPE + base);
#pragma unroll
    for (int r = 0; r < 8; ++r) {
      float pe = acc[r] + sBias[ncol];
      vv[r] += pe;                               // v + pe
      kp[r] = to_bf((float)kp[r] + pe);          // pe - kk
    }
    *(v8f*)(sVPE + base) = vv;
    *(v8bf*)(sKPE + base) = kp;
  }
  __syncthreads();

  // ---- P6: scores s(p,k,h) = mean_d(q + (pe-kk)) ----
  {
    const int row = tid, p = row >> 4;
#pragma unroll
    for (int h = 0; h < NHEAD; ++h) {
      float s = 0.f;
      for (int dd = 0; dd < d; ++dd) {
        int c = h * d + dd;
        s += sQ[p * 64 + c] + (float)sKPE[c * 256 + row];
      }
      sS[row * 4 + h] = s * dinv;
    }
  }
  __syncthreads();

  // ---- P7: softmax statistics per (p,h) ----
  if (tid < 64) {
    const int p = tid >> 2, h = tid & 3;
    float m = -1e30f;
    for (int k = 0; k < KNN; ++k) m = fmaxf(m, sS[(p * 16 + k) * 4 + h]);
    float ssum = 0.f;
    for (int k = 0; k < KNN; ++k) ssum += __expf(sS[(p * 16 + k) * 4 + h] - m);
    sMax[tid] = m;
    sSum[tid] = 1.0f / ssum;
  }
  __syncthreads();

  // ---- P8: normalize to attention weights ----
  {
    const int row = tid, p = row >> 4;
#pragma unroll
    for (int h = 0; h < NHEAD; ++h)
      sS[row * 4 + h] = __expf(sS[row * 4 + h] - sMax[p * 4 + h]) * sSum[p * 4 + h];
  }
  __syncthreads();

  // ---- P9: agg(p,c) = sum_k attn * (v+pe)  (reuse sQ as agg) ----
  for (int e = tid; e < 16 * cout; e += 256) {
    int p = e / cout, c = e % cout, h = c / d;
    float acc = 0.f;
#pragma unroll
    for (int k = 0; k < KNN; ++k)
      acc += sS[(p * 16 + k) * 4 + h] * sVPE[c * 256 + p * 16 + k];
    sQ[p * 64 + c] = acc;
  }
  __syncthreads();

  // ---- P10: out = relu(skip + agg@Wfc + bfc) into 128-wide concat buffer ----
  for (int e = tid; e < 16 * cout; e += 256) {
    int p = e / cout, c = e % cout;
    float o = bfc[c];
    for (int j2 = 0; j2 < cout; ++j2) o += sQ[p * 64 + j2] * Wfc[j2 * cout + c];
    float sc;
    if (has_sc) {
      sc = bsc[c];
      for (int j2 = 0; j2 < cin; ++j2) sc += sXin[p * 32 + j2] * Wsc[j2 * cout + c];
    } else {
      sc = sXin[p * 32 + c];
    }
    xout[((size_t)b * NPTS + n0 + p) * 128 + c] = fmaxf(sc + o, 0.f);
  }
}

// ---------------------------------------------------------------------------
// Kernel 5: global max-pool over N. ReLU outputs are >= 0, so int atomicMax on
// the float bit pattern is exact.
// ---------------------------------------------------------------------------
__global__ void k_zero_out(float* __restrict__ out) {
  int i = blockIdx.x * blockDim.x + threadIdx.x;
  if (i < BATCH * 128) out[i] = 0.f;
}

__global__ __launch_bounds__(128) void k_maxpool(const float* __restrict__ xall,
                                                 float* __restrict__ out) {
  const int b = blockIdx.y;
  const int c = threadIdx.x;
  const int nbase = blockIdx.x * (NPTS / 32);
  float m = 0.f;
  for (int n = 0; n < NPTS / 32; ++n)
    m = fmaxf(m, xall[((size_t)b * NPTS + nbase + n) * 128 + c]);
  atomicMax((int*)&out[b * 128 + c], __float_as_int(m));
}

// ---------------------------------------------------------------------------
extern "C" void kernel_launch(void* const* d_in, const int* in_sizes, int n_in,
                              void* d_out, int out_size, void* d_ws, size_t ws_size,
                              hipStream_t stream) {
  struct LW {
    const float *Wq, *bq, *Wk, *bk, *Wv, *bv, *Wp1, *bp1, *Wp2, *bp2, *Wfc, *bfc, *Wsc, *bsc;
  } L[4];
  const float *x, *inW, *inb;
  auto F = [&](int i) { return (const float*)d_in[i]; };

  if (n_in > 0 && in_sizes[0] == BATCH * 3 * NPTS) {
    // dict insertion order: x, in_W, in_b, L1.., L2.., L3.., L4..
    x = F(0); inW = F(1); inb = F(2);
    int i = 3;
    for (int l = 0; l < 4; ++l) {
      L[l].Wq = F(i++); L[l].bq = F(i++); L[l].Wk = F(i++); L[l].bk = F(i++);
      L[l].Wv = F(i++); L[l].bv = F(i++); L[l].Wp1 = F(i++); L[l].bp1 = F(i++);
      L[l].Wp2 = F(i++); L[l].bp2 = F(i++); L[l].Wfc = F(i++); L[l].bfc = F(i++);
      if (l >= 2) { L[l].Wsc = F(i++); L[l].bsc = F(i++); }
      else        { L[l].Wsc = L[l].Wq; L[l].bsc = L[l].bq; }
    }
  } else {
    // JAX pytree order (keys sorted): L1..L4, in_W, in_b, x
    int i = 0;
    for (int l = 0; l < 4; ++l) {
      L[l].Wfc = F(i++); L[l].Wk = F(i++); L[l].Wp1 = F(i++); L[l].Wp2 = F(i++);
      L[l].Wq  = F(i++);
      if (l >= 2) L[l].Wsc = F(i++);
      L[l].Wv = F(i++);
      L[l].bfc = F(i++); L[l].bk = F(i++); L[l].bp1 = F(i++); L[l].bp2 = F(i++);
      L[l].bq = F(i++);
      if (l >= 2) L[l].bsc = F(i++); else { L[l].Wsc = L[l].Wq; L[l].bsc = L[l].bq; }
      L[l].bv = F(i++);
    }
    inW = F(i++); inb = F(i++); x = F(i++);
  }

  // Workspace carve (~27 MB)
  char* w = (char*)d_ws;
  float* pos = (float*)w;   w += (size_t)BATCH * NPTS * 3 * sizeof(float);
  int* idx = (int*)w;       w += (size_t)BATCH * NPTS * KNN * sizeof(int);
  float* pdiff = (float*)w; w += (size_t)BATCH * NPTS * KNN * 3 * sizeof(float);
  float* f = (float*)w;     w += (size_t)BATCH * NPTS * 16 * sizeof(float);
  float* xall = (float*)w;  w += (size_t)BATCH * NPTS * 128 * sizeof(float);
  float* out = (float*)d_out;

  k_normalize<<<BATCH, 256, 0, stream>>>(x, pos);
  k_knn<<<dim3(NPTS / 256, BATCH), 256, 0, stream>>>(pos, idx, pdiff);
  k_embed<<<(BATCH * NPTS + 255) / 256, 256, 0, stream>>>(pos, inW, inb, f);

  const dim3 lg(NPTS / 16, BATCH);
#define LAUNCH_LAYER(l, XIN, XSTR, CIN, XOUT, COUT, SC)                          \
  k_layer<<<lg, 256, 0, stream>>>(XIN, XSTR, CIN, XOUT, COUT, SC, idx, pdiff,    \
      L[l].Wq, L[l].bq, L[l].Wk, L[l].bk, L[l].Wv, L[l].bv, L[l].Wp1, L[l].bp1,  \
      L[l].Wp2, L[l].bp2, L[l].Wfc, L[l].bfc, L[l].Wsc, L[l].bsc)

  LAUNCH_LAYER(0, f,         16, 16, xall + 0,  16, 0);
  LAUNCH_LAYER(1, xall + 0,  128, 16, xall + 16, 16, 0);
  LAUNCH_LAYER(2, xall + 16, 128, 16, xall + 32, 32, 1);
  LAUNCH_LAYER(3, xall + 32, 128, 32, xall + 64, 64, 1);
#undef LAUNCH_LAYER

  k_zero_out<<<(BATCH * 128 + 255) / 256, 256, 0, stream>>>(out);
  k_maxpool<<<dim3(32, BATCH), 128, 0, stream>>>(xall, out);
}
